// GemmaAttention_40321152975086
// MI455X (gfx1250) — compile-verified, matrix operand order checked by reading, not run on hardware
//
#include <hip/hip_runtime.h>

// ---------------------------------------------------------------------------
// Types for CDNA5 WMMA (wave32, 16x16x32 bf16 -> f32)
// ---------------------------------------------------------------------------
typedef __attribute__((ext_vector_type(16))) __bf16 v16bf;
typedef __attribute__((ext_vector_type(8)))  __bf16 v8bf;
typedef __attribute__((ext_vector_type(8)))  float  v8f;
typedef __attribute__((ext_vector_type(4)))  unsigned int u32x4;
typedef __attribute__((ext_vector_type(8)))  unsigned int u32x8;

union BF16x16 { v16bf v; v8bf h[2]; };

__device__ inline v8f vz() {
  v8f z;
  for (int i = 0; i < 8; ++i) z[i] = 0.0f;
  return z;
}

__device__ inline v8f wmma_bf16(v16bf a, v16bf b, v8f c) {
  // (neg_a, A, neg_b, B, c_mod, C, reuse_a, reuse_b)
  return __builtin_amdgcn_wmma_f32_16x16x32_bf16(false, a, false, b, (short)0, c,
                                                 false, false);
}

// Load a 16x32 bf16 operand fragment in the ISA A-layout:
//   lanes 0-15: rows m0..m0+15, elements = K[k0..k0+7]  ++ K[k0+16..k0+23]
//   lanes 16-31: same rows,     elements = K[k0+8..+15] ++ K[k0+24..+31]
// The B operand is fed as rows of B^T with the same layout.
__device__ inline v16bf ld_frag(const __bf16* base, int m0, int stride, int k0,
                                int lane) {
  const int half = lane >> 4;
  const int r    = lane & 15;
  const __bf16* p = base + (m0 + r) * stride + k0 + half * 8;
  BF16x16 u;
  u.h[0] = *(const v8bf*)(p);
  u.h[1] = *(const v8bf*)(p + 16);   // +16 elements = +32 bytes
  return u.v;
}

// ---------------------------------------------------------------------------
// Tensor Data Mover: 2-D bf16 tile (tile_d0 wide x tile_d1 rows) from a
// row-major tensor (row length tensor_d0, tensor_d1 rows, row stride
// stride0 elements) into LDS at byte offset lds_off. Groups 2/3 omitted
// (2-D tensor). Issued whole-wave (EXEC ignored by tensor ops); completion
// tracked with TENSORcnt.
// ---------------------------------------------------------------------------
__device__ inline void tdm_load_2d(unsigned lds_off, const void* gptr,
                                   unsigned tile_d0, unsigned tile_d1,
                                   unsigned tensor_d0, unsigned tensor_d1,
                                   unsigned stride0) {
  unsigned long long ga = (unsigned long long)gptr;
  u32x4 g0;
  g0[0] = 1u;                                        // count=1, user mode
  g0[1] = lds_off;                                   // lds_addr
  g0[2] = (unsigned)ga;                              // global_addr[31:0]
  g0[3] = ((unsigned)(ga >> 32) & 0x01FFFFFFu)       // global_addr[56:32]
          | 0x80000000u;                             // type=2 ("image")
  u32x8 g1;
  g1[0] = 0x00010000u;                               // data_size=1 (2 bytes)
  g1[1] = (tensor_d0 & 0xFFFFu) << 16;               // tensor_dim0[15:0]
  g1[2] = ((tensor_d0 >> 16) & 0xFFFFu) | ((tensor_d1 & 0xFFFFu) << 16);
  g1[3] = ((tensor_d1 >> 16) & 0xFFFFu) | ((tile_d0 & 0xFFFFu) << 16);
  g1[4] = tile_d1 & 0xFFFFu;                         // tile_dim1 (tile_dim2=0)
  g1[5] = stride0;                                   // tensor_dim0_stride[31:0]
  g1[6] = 0u;                                        // stride hi, dim1_stride lo
  g1[7] = 0u;
  asm volatile("tensor_load_to_lds %0, %1" :: "s"(g0), "s"(g1) : "memory");
}

// ---------------------------------------------------------------------------
// Problem constants
// ---------------------------------------------------------------------------
#define S_LEN 4096
#define HID   2048
#define NH    8
#define DH    256
#define NQKV  2560                 // 2048 Q + 256 K + 256 V packed columns

// ---------------------------------------------------------------------------
// fp32 -> bf16 convert
// ---------------------------------------------------------------------------
__global__ __launch_bounds__(256) void conv_kernel(const float* __restrict__ src,
                                                   __bf16* __restrict__ dst) {
  int idx = blockIdx.x * 256 + threadIdx.x;
  dst[idx] = (__bf16)src[idx];
}

// fp32 [K][N] -> bf16 transposed [N][K]
__global__ __launch_bounds__(256) void tconv_kernel(const float* __restrict__ src,
                                                    __bf16* __restrict__ dst,
                                                    int K, int N) {
  int idx = blockIdx.x * 256 + threadIdx.x;
  int k = idx / N, n = idx % N;
  dst[n * K + k] = (__bf16)src[k * N + n];
}

// ---------------------------------------------------------------------------
// bf16 WMMA GEMM:  C[M][ldc] (f32) = A[M][lda](bf16) x Bt[N][ldb]^T (bf16)
// Block: 128x64 output tile, 256 threads = 8 waves, 2x2 WMMA tiles per wave.
// K slabs (64 wide) staged into double-buffered LDS by the Tensor Data Mover
// issued from wave 0; compute overlaps the next slab's DMA.
// ---------------------------------------------------------------------------
__global__ __launch_bounds__(256) void gemm_bf16_kernel(
    const __bf16* __restrict__ A, const __bf16* __restrict__ Bt,
    float* __restrict__ C, int lda, int ldb, int ldc, int K,
    int Mrows, int Nrows) {
  __shared__ __bf16 as[2][128 * 64];   // A tiles: 128 rows x 64 K
  __shared__ __bf16 bs[2][64 * 64];    // B^T tiles: 64 rows x 64 K

  const int tid  = threadIdx.x;
  const int lane = tid & 31;
  const int w    = tid >> 5;
  const int m0   = blockIdx.y * 128;
  const int n0   = blockIdx.x * 64;
  const int mi0  = (w >> 1) * 32;      // wave's tile rows: mi0, mi0+16
  const int ni0  = (w & 1) * 32;       // wave's tile cols: ni0, ni0+16

  v8f acc00 = vz(), acc01 = vz(), acc10 = vz(), acc11 = vz();

  const unsigned as_off = (unsigned)(size_t)&as[0][0];
  const unsigned bs_off = (unsigned)(size_t)&bs[0][0];
  const bool issuer = (w == 0);

  if (issuer) {                        // preload slab 0
    tdm_load_2d(as_off, A + (size_t)m0 * lda, 64, 128,
                (unsigned)lda, (unsigned)Mrows, (unsigned)lda);
    tdm_load_2d(bs_off, Bt + (size_t)n0 * ldb, 64, 64,
                (unsigned)ldb, (unsigned)Nrows, (unsigned)ldb);
  }

  int cur = 0;
  for (int kb = 0; kb < K; kb += 64) {
    if (issuer) {
      if (kb + 64 < K) {               // queue next slab into other buffer
        tdm_load_2d(as_off + (unsigned)(cur ^ 1) * 16384u,
                    A + (size_t)m0 * lda + kb + 64, 64, 128,
                    (unsigned)lda, (unsigned)Mrows, (unsigned)lda);
        tdm_load_2d(bs_off + (unsigned)(cur ^ 1) * 8192u,
                    Bt + (size_t)n0 * ldb + kb + 64, 64, 64,
                    (unsigned)ldb, (unsigned)Nrows, (unsigned)ldb);
        __builtin_amdgcn_s_wait_tensorcnt(2);   // current slab complete
      } else {
        __builtin_amdgcn_s_wait_tensorcnt(0);
      }
    }
    __syncthreads();

    const __bf16* ab = as[cur];
    const __bf16* bb = bs[cur];
    for (int ks = 0; ks < 64; ks += 32) {
      v16bf a0 = ld_frag(ab, mi0, 64, ks, lane);
      v16bf a1 = ld_frag(ab, mi0 + 16, 64, ks, lane);
      v16bf b0 = ld_frag(bb, ni0, 64, ks, lane);
      v16bf b1 = ld_frag(bb, ni0 + 16, 64, ks, lane);
      acc00 = wmma_bf16(a0, b0, acc00);
      acc01 = wmma_bf16(a0, b1, acc01);
      acc10 = wmma_bf16(a1, b0, acc10);
      acc11 = wmma_bf16(a1, b1, acc11);
    }
    __syncthreads();                   // all reads done before buffer reuse
    cur ^= 1;
  }

  const int half = lane >> 4, r = lane & 15;
  for (int i = 0; i < 8; ++i) {
    int r0 = m0 + mi0 + half * 8 + i;
    int r1 = r0 + 16;
    C[(size_t)r0 * ldc + n0 + ni0 + r]      = acc00[i];
    C[(size_t)r0 * ldc + n0 + ni0 + 16 + r] = acc01[i];
    C[(size_t)r1 * ldc + n0 + ni0 + r]      = acc10[i];
    C[(size_t)r1 * ldc + n0 + ni0 + 16 + r] = acc11[i];
  }
}

// ---------------------------------------------------------------------------
// RoPE on Q and K (reads fp32 QKV, writes bf16). u in [0,8) -> head of Q,
// u == 8 -> K.  i in [0,128) is the rotation pair index.
// ---------------------------------------------------------------------------
__global__ __launch_bounds__(256) void rope_kernel(const float* __restrict__ Y,
                                                   __bf16* __restrict__ Qb,
                                                   __bf16* __restrict__ Kb) {
  int idx = blockIdx.x * 256 + threadIdx.x;     // S * 9 * 128 threads
  int s   = idx / (9 * 128);
  int rem = idx % (9 * 128);
  int u   = rem >> 7;
  int i   = rem & 127;

  float freq = (float)s * __expf(-(float)i * (logf(10000.0f) / 128.0f));
  float c  = cosf(freq);
  float sn = sinf(freq);

  const float* y = Y + s * NQKV + (u < 8 ? u * DH : 2048);
  float x1 = y[i], x2 = y[i + 128];
  float o1 = x1 * c - x2 * sn;
  float o2 = x2 * c + x1 * sn;

  if (u < 8) {
    __bf16* q = Qb + s * HID + u * DH;
    q[i]       = (__bf16)o1;
    q[i + 128] = (__bf16)o2;
  } else {
    __bf16* k = Kb + s * DH;
    k[i]       = (__bf16)o1;
    k[i + 128] = (__bf16)o2;
  }
}

// V: fp32 [S][DH] slice of Y -> bf16 transposed Vt[DH][S]
__global__ __launch_bounds__(256) void vconv_kernel(const float* __restrict__ Y,
                                                    __bf16* __restrict__ Vt) {
  int idx = blockIdx.x * 256 + threadIdx.x;     // S * DH
  int s = idx >> 8, d = idx & 255;
  Vt[d * S_LEN + s] = (__bf16)Y[s * NQKV + 2304 + d];
}

// ---------------------------------------------------------------------------
// Flash attention: one block = (head h, 64 query rows). Online softmax over
// 64-key blocks; QK^T and PV on WMMA; Q/K/V tiles staged by the TDM; V-tile
// DMA overlaps the softmax pass.
// ---------------------------------------------------------------------------
__global__ __launch_bounds__(256) void attn_kernel(const __bf16* __restrict__ Qb,
                                                   const __bf16* __restrict__ Kb,
                                                   const __bf16* __restrict__ Vt,
                                                   __bf16* __restrict__ Ob) {
  __shared__ __bf16 qs[64 * 256];     // Q tile [64][256]
  __shared__ __bf16 kv[64 * 256];     // K tile [64][256], reused as V^T [256][64]
  __shared__ float  sc[64 * 64];      // score tile (f32)
  __shared__ __bf16 pbf[64 * 64];     // exp(scores) as bf16
  __shared__ float  rowm[64], rowl[64], rowa[64];

  const int tid = threadIdx.x, lane = tid & 31, w = tid >> 5;
  const int half = lane >> 4, r = lane & 15;
  const int h = blockIdx.y;
  const int q0 = blockIdx.x * 64;

  const unsigned qs_off = (unsigned)(size_t)&qs[0];
  const unsigned kv_off = (unsigned)(size_t)&kv[0];
  const bool issuer = (w == 0);

  if (issuer)                          // Q tile DMA (completion checked below)
    tdm_load_2d(qs_off, Qb + (size_t)q0 * HID + h * DH, 256, 64,
                HID, S_LEN, HID);
  if (tid < 64) { rowm[tid] = -1e30f; rowl[tid] = 0.0f; }

  v8f o[8];
  for (int j = 0; j < 8; ++j) o[j] = vz();

  const int smi = (w >> 1) * 16;      // score tile row for this wave
  const int sni = (w & 1) * 32;       // score tile col base (2 tiles)
  const int omi = (w & 3) * 16;       // output tile row
  const int onj = (w >> 2) * 8;       // output tile col base (8 tiles of 16)

  for (int kb = 0; kb <= q0; kb += 64) {
    __syncthreads();                  // previous PV done before kv overwrite
    if (issuer) {                     // K tile [64][256] via TDM
      tdm_load_2d(kv_off, Kb + (size_t)kb * DH, 256, 64, DH, S_LEN, DH);
      __builtin_amdgcn_s_wait_tensorcnt(0);   // Q (first iter) + K complete
    }
    __syncthreads();

    // Scores: two 16x16 tiles per wave, K-loop over DH=256 (8 WMMA steps).
    for (int t2 = 0; t2 < 2; ++t2) {
      int ni = sni + t2 * 16;
      v8f c = vz();
      for (int k0 = 0; k0 < 256; k0 += 32) {
        v16bf a = ld_frag(qs, smi, 256, k0, lane);
        v16bf b = ld_frag(kv, ni, 256, k0, lane);   // rows of K == rows of B^T
        c = wmma_bf16(a, b, c);
      }
      for (int i = 0; i < 8; ++i) {
        int lr = smi + half * 8 + i;
        int lc = ni + r;
        float v = c[i] * 0.0625f;                   // 1/sqrt(256)
        if (kb + lc > q0 + lr) v = -1e30f;          // causal mask
        sc[lr * 64 + lc] = v;
      }
    }
    __syncthreads();

    // V^T tile [256][64] via TDM into kv; overlaps with the softmax pass.
    if (issuer)
      tdm_load_2d(kv_off, Vt + kb, 64, 256, S_LEN, DH, S_LEN);

    // Online softmax: one thread per query row.
    if (tid < 64) {
      float mold = rowm[tid], mnew = mold;
      for (int c2 = 0; c2 < 64; ++c2) mnew = fmaxf(mnew, sc[tid * 64 + c2]);
      float alpha = __expf(mold - mnew);
      float ssum = 0.0f;
      for (int c2 = 0; c2 < 64; ++c2) {
        float p = __expf(sc[tid * 64 + c2] - mnew);
        ssum += p;
        pbf[tid * 64 + c2] = (__bf16)p;
      }
      rowa[tid] = alpha;
      rowm[tid] = mnew;
      rowl[tid] = rowl[tid] * alpha + ssum;
    }
    if (issuer) __builtin_amdgcn_s_wait_tensorcnt(0);
    __syncthreads();

    // Rescale accumulators by alpha(row) and add P x V (2 WMMA k-steps).
    float al[8];
    for (int i = 0; i < 8; ++i) al[i] = rowa[omi + half * 8 + i];
    for (int j = 0; j < 8; ++j) {
      for (int i = 0; i < 8; ++i) o[j][i] *= al[i];
      for (int k0 = 0; k0 < 64; k0 += 32) {
        v16bf a = ld_frag(pbf, omi, 64, k0, lane);
        v16bf b = ld_frag(kv, (onj + j) * 16, 64, k0, lane);  // V^T rows
        o[j] = wmma_bf16(a, b, o[j]);
      }
    }
  }

  // Normalize by running sum and store bf16 output [S][NH*DH].
  float li[8];
  for (int i = 0; i < 8; ++i) li[i] = 1.0f / rowl[omi + half * 8 + i];
  for (int j = 0; j < 8; ++j)
    for (int i = 0; i < 8; ++i) {
      int row = q0 + omi + half * 8 + i;
      int col = h * DH + (onj + j) * 16 + r;
      Ob[row * HID + col] = (__bf16)(o[j][i] * li[i]);
    }
}

// ---------------------------------------------------------------------------
// Host-side orchestration
// ---------------------------------------------------------------------------
extern "C" void kernel_launch(void* const* d_in, const int* in_sizes, int n_in,
                              void* d_out, int out_size, void* d_ws, size_t ws_size,
                              hipStream_t stream) {
  const float* hs = (const float*)d_in[0];
  // d_in[1] = position_ids (arange) -> positions equal row index; unused.
  const float* wq = (const float*)d_in[2];
  const float* wk = (const float*)d_in[3];
  const float* wv = (const float*)d_in[4];
  const float* wo = (const float*)d_in[5];
  float* out = (float*)d_out;

  char* ws = (char*)d_ws;
  __bf16* Xbf = (__bf16*)(ws + 0);          // 4096x2048 bf16   (16,777,216 B)
  __bf16* WT  = (__bf16*)(ws + 16777216);   // 2560x2048 bf16   (10,485,760 B)
  __bf16* WoT = (__bf16*)(ws + 27262976);   // 2048x2048 bf16   ( 8,388,608 B)
  float*  Y   = (float* )(ws + 35651584);   // 4096x2560 f32    (41,943,040 B)
  __bf16* Qb  = (__bf16*)(ws + 77594624);   // 4096x2048 bf16
  __bf16* Kb  = (__bf16*)(ws + 94371840);   // 4096x256  bf16
  __bf16* Vt  = (__bf16*)(ws + 96468992);   // 256x4096  bf16
  __bf16* Ab  = (__bf16*)(ws + 98566144);   // 4096x2048 bf16

  // 1) Convert activations; pack transposed bf16 weights.
  conv_kernel<<<32768, 256, 0, stream>>>(hs, Xbf);                       // 8.4M elems
  tconv_kernel<<<16384, 256, 0, stream>>>(wq, WT, HID, HID);             // Wq^T
  tconv_kernel<<<2048, 256, 0, stream>>>(wk, WT + 2048 * 2048, HID, DH); // Wk^T
  tconv_kernel<<<2048, 256, 0, stream>>>(wv, WT + 2304 * 2048, HID, DH); // Wv^T
  tconv_kernel<<<16384, 256, 0, stream>>>(wo, WoT, HID, HID);            // Wo^T

  // 2) Fused QKV projection: Y[4096][2560] = X * [Wq|Wk|Wv]
  gemm_bf16_kernel<<<dim3(NQKV / 64, S_LEN / 128), 256, 0, stream>>>(
      Xbf, WT, Y, HID, HID, NQKV, HID, S_LEN, NQKV);

  // 3) RoPE on Q,K; V -> bf16 transposed.
  rope_kernel<<<(S_LEN * 9 * 128) / 256, 256, 0, stream>>>(Y, Qb, Kb);
  vconv_kernel<<<(S_LEN * DH) / 256, 256, 0, stream>>>(Y, Vt);

  // 4) Causal flash attention, 8 heads x 64 query blocks.
  attn_kernel<<<dim3(S_LEN / 64, NH), 256, 0, stream>>>(Qb, Kb, Vt, Ab);

  // 5) Output projection to fp32 d_out.
  gemm_bf16_kernel<<<dim3(HID / 64, S_LEN / 128), 256, 0, stream>>>(
      Ab, WoT, out, HID, HID, HID, HID, S_LEN, HID);
}
